// SelfAttention_23098334118063
// MI455X (gfx1250) — matrix operand order repaired and checked by reading, not compiled, not run on hardware
//
#include <hip/hip_runtime.h>

// ---------------------------------------------------------------------------
// MI455X self-attention (B=4, C=512, H=W=64 -> N=4096, Cqk=64)
// Compute-bound (~99 GFLOP vs ~100MB HBM): v_wmma_f32_16x16x32_f16 everywhere.
// Flash-attention: never materialize the 4096x4096 energy matrix.
// K/V/Q tiles staged with GLOBAL_LOAD_ASYNC_TO_LDS_B128 (ASYNCcnt path).
// ---------------------------------------------------------------------------

typedef __attribute__((ext_vector_type(16))) _Float16 v16h;
typedef __attribute__((ext_vector_type(8)))  float    v8f;

#define NSP 4096   // spatial positions H*W
#define CIN 512    // channels

// Async 16B-per-lane copy global -> LDS, tracked by ASYNCcnt (no VGPR round trip)
__device__ __forceinline__ void async_b128(unsigned lds_byte_off, const void* gaddr) {
    asm volatile("global_load_async_to_lds_b128 %0, %1, off"
                 :: "v"(lds_byte_off), "v"(gaddr)
                 : "memory");
}
__device__ __forceinline__ void wait_async0() {
    asm volatile("s_wait_asynccnt 0" ::: "memory");
}

// ---------------------------------------------------------------------------
// Projection GEMM: out[n, o] = sum_c W[o,c] * X[b,c,n] + bias[o], f16 output.
// Workgroup = 8 waves computing a 64(n) x 32(o) tile; K-loop over c in 32s.
// Staging: float4 global loads into registers (clause + single wait), then
// convert + transpose-store into LDS.
// ---------------------------------------------------------------------------
__global__ void __launch_bounds__(256)
proj_kernel(const float* __restrict__ X,    // (B, CIN, NSP)
            const float* __restrict__ W,    // (O, CIN)
            const float* __restrict__ bias, // (O)
            _Float16* __restrict__ out,
            int rowStride, int colStride, long long batchStride)
{
    __shared__ _Float16 XT[64 * 40];  // [n_local][c_local], pad->40 (80B rows)
    __shared__ _Float16 WT[32 * 40];  // [o_local][c_local]

    const int tid  = threadIdx.x;
    const int lane = tid & 31;
    const int wid  = tid >> 5;
    const int wn   = wid & 3;   // n sub-tile (4 x 16 rows)
    const int wo   = wid >> 2;  // o sub-tile (2 x 16 cols)
    const int b    = blockIdx.z;
    const int n0   = blockIdx.x * 64;
    const int o0   = blockIdx.y * 32;

    v8f acc = {};

    for (int c0 = 0; c0 < CIN; c0 += 32) {
        // ---- batched vector loads first (one wait), then convert/store
        float4 xr[2];
        #pragma unroll
        for (int it = 0; it < 2; ++it) {
            int idx = tid + it * 256;            // 512 float4s = 32c x 64n
            int cl = idx >> 4, nl4 = (idx & 15) * 4;
            xr[it] = *(const float4*)&X[((size_t)(b * CIN + c0 + cl)) * NSP + n0 + nl4];
        }
        const int wol = tid >> 3, wcl4 = (tid & 7) * 4;  // 256 float4s = 32o x 32c
        float4 wr = *(const float4*)&W[(size_t)(o0 + wol) * CIN + c0 + wcl4];

        #pragma unroll
        for (int it = 0; it < 2; ++it) {
            int idx = tid + it * 256;
            int cl = idx >> 4, nl4 = (idx & 15) * 4;
            XT[(nl4 + 0) * 40 + cl] = (_Float16)xr[it].x;
            XT[(nl4 + 1) * 40 + cl] = (_Float16)xr[it].y;
            XT[(nl4 + 2) * 40 + cl] = (_Float16)xr[it].z;
            XT[(nl4 + 3) * 40 + cl] = (_Float16)xr[it].w;
        }
        {
            union { unsigned long long u; _Float16 h[4]; } pk;
            pk.h[0] = (_Float16)wr.x; pk.h[1] = (_Float16)wr.y;
            pk.h[2] = (_Float16)wr.z; pk.h[3] = (_Float16)wr.w;
            *(unsigned long long*)&WT[wol * 40 + wcl4] = pk.u;  // one ds_store_b64
        }
        __syncthreads();

        // A fragment: lane m = lane%16, kg = lane/16; j<8 -> k=kg*8+j, j>=8 -> k=16+kg*8+(j-8)
        union { int4 q[2]; v16h v; } af, bf;
        const int arow = (wn * 16 + (lane & 15)) * 40;
        const int kg   = lane >> 4;
        const int4* ap = (const int4*)&XT[arow + kg * 8];
        af.q[0] = ap[0];
        af.q[1] = ap[2];  // +16 halves
        const int4* bp = (const int4*)&WT[(wo * 16 + (lane & 15)) * 40 + kg * 8];
        bf.q[0] = bp[0];
        bf.q[1] = bp[2];

        acc = __builtin_amdgcn_wmma_f32_16x16x32_f16(
                  false, af.v, false, bf.v, (short)0, acc, false, false);
        __syncthreads();
    }

    // C/D layout: lane -> N(col), VGPR r -> M(row) (+8 for lanes 16-31)
    const int ocol    = o0 + wo * 16 + (lane & 15);
    const float bb    = bias[ocol];
    const int rowbase = n0 + wn * 16 + ((lane >> 4) << 3);
    if (rowStride == 1) {
        // v layout (B,512,N): 8 rows are contiguous halves -> one b128 store
        union { int4 q; _Float16 h[8]; } pk;
        #pragma unroll
        for (int r = 0; r < 8; ++r) pk.h[r] = (_Float16)(acc[r] + bb);
        *(int4*)&out[(long long)b * batchStride + (long long)ocol * colStride + rowbase] = pk.q;
    } else {
        #pragma unroll
        for (int r = 0; r < 8; ++r) {
            long long n = rowbase + r;
            out[(long long)b * batchStride + n * rowStride + (long long)ocol * colStride] =
                (_Float16)(acc[r] + bb);
        }
    }
}

// ---------------------------------------------------------------------------
// Flash attention: workgroup = 8 waves, 32 query rows, loop over 64-key blocks.
//   S  = Q(32x64) @ K^T       -> 8 waves x 1 tile  x 2 wmma
//   softmax (online max/sum, rescale acc)
//   O += P(32x64) @ V(64x512) -> 8 waves x 8 tiles x 2 wmma (acc in 64 VGPRs)
// Q,K stored (B,N,64) f16; V stored (B,512,N) f16 so LDS is [c][n] and
// B-fragments (k = n contiguous) are 16B-aligned ds_load chunks.
// LDS byte offsets (dynamic shared base == 0, no static LDS here):
//   Qs@0 (4608B) | Ks@4608 (9216B) | Vs@13824 (73728B) | Ps@87552 | Ss@92160...
// ---------------------------------------------------------------------------
#define LDS_Q_OFF 0u
#define LDS_K_OFF 4608u
#define LDS_V_OFF 13824u

__global__ void __launch_bounds__(256)
attn_kernel(const _Float16* __restrict__ Qg,  // (B, NSP, 64)
            const _Float16* __restrict__ Kg,  // (B, NSP, 64)
            const _Float16* __restrict__ Vg,  // (B, CIN, NSP)
            const float* __restrict__ Xg,     // (B, CIN, NSP) residual
            const float* __restrict__ gamma,  // (1)
            float* __restrict__ Og)           // (B, CIN, NSP)
{
    extern __shared__ char smem[];
    _Float16* Qs = (_Float16*)smem;      // 32  x 72
    _Float16* Ks = Qs + 32 * 72;         // 64  x 72
    _Float16* Vs = Ks + 64 * 72;         // 512 x 72
    _Float16* Ps = Vs + 512 * 72;        // 32  x 72
    float*    Ss = (float*)(Ps + 32 * 72);  // 32 x 66
    float*    mst = Ss + 32 * 66;           // 32 running max
    float*    lst = mst + 32;               // 32 running sum
    float*    sst = lst + 32;               // 32 rescale factor
    float*    red = sst + 32;               // 256 reduction scratch

    const int tid  = threadIdx.x;
    const int lane = tid & 31;
    const int wid  = tid >> 5;
    const int b    = blockIdx.y;
    const int n0   = blockIdx.x * 32;     // query tile base
    const int jg   = wid & 1;             // query row group (2 x 16)
    const int ng   = wid >> 1;            // S key-col group (4 x 16)
    const int cg   = wid >> 1;            // PV channel group (4 x 128)

    // stage Q tile asynchronously: 32 rows x 64 halves = 256 x 16B, one per thread
    {
        int j = tid >> 3, d8 = tid & 7;
        async_b128(LDS_Q_OFF + (unsigned)((j * 72 + d8 * 8) * 2),
                   Qg + ((size_t)b * NSP + n0 + j) * 64 + d8 * 8);
    }
    if (tid < 32) { mst[tid] = -3e38f; lst[tid] = 0.f; }
    v8f acc[8];
    #pragma unroll
    for (int t = 0; t < 8; ++t) acc[t] = (v8f){};
    __syncthreads();   // Q landing covered by per-iteration wait_async0 below

    for (int kb = 0; kb < NSP; kb += 64) {
        // ---- async stage K block (64 x 64 halves)
        #pragma unroll
        for (int it = 0; it < 2; ++it) {
            int idx = tid + it * 256;
            int nl = idx >> 3, d8 = idx & 7;
            async_b128(LDS_K_OFF + (unsigned)((nl * 72 + d8 * 8) * 2),
                       Kg + ((size_t)b * NSP + kb + nl) * 64 + d8 * 8);
        }
        // ---- async stage V block (512 c x 64 n halves), rows contiguous in n
        #pragma unroll
        for (int it = 0; it < 16; ++it) {
            int idx = tid + it * 256;
            int c = idx >> 3, n8 = idx & 7;
            async_b128(LDS_V_OFF + (unsigned)((c * 72 + n8 * 8) * 2),
                       Vg + ((size_t)(b * CIN + c)) * NSP + kb + n8 * 8);
        }
        // prefetch next V block into GL2 while we compute this one
        if (kb + 64 < NSP) {
            __builtin_prefetch(Vg + ((size_t)(b * CIN + tid * 2)) * NSP + kb + 64, 0, 0);
            __builtin_prefetch(Vg + ((size_t)(b * CIN + tid * 2 + 1)) * NSP + kb + 64, 0, 0);
        }
        wait_async0();
        __syncthreads();

        // ---- S = Q K^T : one 16x16 tile per wave, K-dim 64 = 2 wmma
        {
            v8f s = {};
            const int arow = (jg * 16 + (lane & 15)) * 72;
            const int brow = (ng * 16 + (lane & 15)) * 72;
            const int kg   = lane >> 4;
            #pragma unroll
            for (int ks = 0; ks < 2; ++ks) {
                union { int4 q[2]; v16h v; } af, bf;
                const int4* ap = (const int4*)&Qs[arow + ks * 32 + kg * 8];
                af.q[0] = ap[0]; af.q[1] = ap[2];
                const int4* bp = (const int4*)&Ks[brow + ks * 32 + kg * 8];
                bf.q[0] = bp[0]; bf.q[1] = bp[2];
                s = __builtin_amdgcn_wmma_f32_16x16x32_f16(
                        false, af.v, false, bf.v, (short)0, s, false, false);
            }
            const int row0 = jg * 16 + ((lane >> 4) << 3);
            const int col  = ng * 16 + (lane & 15);
            #pragma unroll
            for (int r = 0; r < 8; ++r) Ss[(row0 + r) * 66 + col] = s[r];
        }
        __syncthreads();

        // ---- online softmax: 8 threads per row over 64 cols
        const int srow = tid >> 3, sc8 = (tid & 7) * 8;
        {
            float pm = -3e38f;
            #pragma unroll
            for (int u = 0; u < 8; ++u) pm = fmaxf(pm, Ss[srow * 66 + sc8 + u]);
            red[tid] = pm;
        }
        __syncthreads();
        if (tid < 32) {
            float mb = red[tid * 8];
            #pragma unroll
            for (int u = 1; u < 8; ++u) mb = fmaxf(mb, red[tid * 8 + u]);
            float mo = mst[tid];
            float mn = fmaxf(mo, mb);
            float sc = __expf(mo - mn);   // first iter: exp(-inf) -> 0
            mst[tid] = mn; sst[tid] = sc; lst[tid] *= sc;
        }
        __syncthreads();
        {
            float mrow = mst[srow];
            float psum = 0.f;
            #pragma unroll
            for (int u = 0; u < 8; ++u) {
                float p = __expf(Ss[srow * 66 + sc8 + u] - mrow);
                Ps[srow * 72 + sc8 + u] = (_Float16)p;
                psum += p;
            }
            red[tid] = psum;
        }
        __syncthreads();
        if (tid < 32) {
            float s8 = 0.f;
            #pragma unroll
            for (int u = 0; u < 8; ++u) s8 += red[tid * 8 + u];
            lst[tid] += s8;
        }

        // ---- rescale accumulators, then O += P @ V
        {
            const int rowbase = jg * 16 + ((lane >> 4) << 3);
            float scr[8];
            #pragma unroll
            for (int r = 0; r < 8; ++r) scr[r] = sst[rowbase + r];
            #pragma unroll
            for (int t = 0; t < 8; ++t)
                #pragma unroll
                for (int r = 0; r < 8; ++r) acc[t][r] *= scr[r];

            const int arow = (jg * 16 + (lane & 15)) * 72;
            const int kg   = lane >> 4;
            #pragma unroll
            for (int ks = 0; ks < 2; ++ks) {
                union { int4 q[2]; v16h v; } af;
                const int4* ap = (const int4*)&Ps[arow + ks * 32 + kg * 8];
                af.q[0] = ap[0]; af.q[1] = ap[2];
                #pragma unroll
                for (int t = 0; t < 8; ++t) {
                    const int ccol = cg * 128 + t * 16 + (lane & 15);
                    union { int4 q[2]; v16h v; } bf;
                    const int4* bp = (const int4*)&Vs[ccol * 72 + ks * 32 + kg * 8];
                    bf.q[0] = bp[0]; bf.q[1] = bp[2];
                    acc[t] = __builtin_amdgcn_wmma_f32_16x16x32_f16(
                                 false, af.v, false, bf.v, (short)0, acc[t], false, false);
                }
            }
        }
        __syncthreads();
    }

    // ---- epilogue: out = gamma * acc / l + x  (8 rows = 8 contiguous floats)
    const float g = gamma[0];
    const int rowbase = jg * 16 + ((lane >> 4) << 3);
    float rinv[8];
    #pragma unroll
    for (int r = 0; r < 8; ++r) rinv[r] = 1.0f / lst[rowbase + r];
    const int j0 = n0 + rowbase;
    #pragma unroll
    for (int t = 0; t < 8; ++t) {
        const int ccol = cg * 128 + t * 16 + (lane & 15);
        size_t base = ((size_t)(b * CIN + ccol)) * NSP + j0;
        float4 xlo = *(const float4*)&Xg[base];
        float4 xhi = *(const float4*)&Xg[base + 4];
        float4 olo, ohi;
        olo.x = g * acc[t][0] * rinv[0] + xlo.x;
        olo.y = g * acc[t][1] * rinv[1] + xlo.y;
        olo.z = g * acc[t][2] * rinv[2] + xlo.z;
        olo.w = g * acc[t][3] * rinv[3] + xlo.w;
        ohi.x = g * acc[t][4] * rinv[4] + xhi.x;
        ohi.y = g * acc[t][5] * rinv[5] + xhi.y;
        ohi.z = g * acc[t][6] * rinv[6] + xhi.z;
        ohi.w = g * acc[t][7] * rinv[7] + xhi.w;
        *(float4*)&Og[base]     = olo;
        *(float4*)&Og[base + 4] = ohi;
    }
}

// ---------------------------------------------------------------------------
extern "C" void kernel_launch(void* const* d_in, const int* in_sizes, int n_in,
                              void* d_out, int out_size, void* d_ws, size_t ws_size,
                              hipStream_t stream) {
    const float* x     = (const float*)d_in[0];
    const float* cw    = (const float*)d_in[1];
    const float* Wq    = (const float*)d_in[2];
    const float* bq    = (const float*)d_in[3];
    const float* Wk    = (const float*)d_in[4];
    const float* bk    = (const float*)d_in[5];
    const float* Wv    = (const float*)d_in[6];
    const float* bv    = (const float*)d_in[7];
    const float* gamma = (const float*)d_in[8];
    float* out = (float*)d_out;

    // workspace: q (2MB) | k (2MB) | v (16MB), all f16
    _Float16* qws = (_Float16*)d_ws;
    _Float16* kws = qws + (size_t)4 * NSP * 64;
    _Float16* vws = kws + (size_t)4 * NSP * 64;

    dim3 blk(256);
    // q: (B,N,64) row-major   k: (B,N,64) row-major   v: (B,512,N)
    proj_kernel<<<dim3(64, 2, 4),  blk, 0, stream>>>(cw, Wq, bq, qws, 64, 1, (long long)NSP * 64);
    proj_kernel<<<dim3(64, 2, 4),  blk, 0, stream>>>(x,  Wk, bk, kws, 64, 1, (long long)NSP * 64);
    proj_kernel<<<dim3(64, 16, 4), blk, 0, stream>>>(x,  Wv, bv, vws, 1, NSP, (long long)CIN * NSP);

    // dynamic LDS: (32+64+512+32)*72 halves + (32*66 + 96 + 256) floats = 102016 B
    const size_t lds_bytes = (size_t)(32 + 64 + 512 + 32) * 72 * 2
                           + (size_t)(32 * 66 + 32 * 3 + 256) * 4;
    attn_kernel<<<dim3(NSP / 32, 4), blk, lds_bytes, stream>>>(qws, kws, vws, x, gamma, out);
}
